// SpinDistanceEdge_7816840478893
// MI455X (gfx1250) — matrix operand order repaired and checked by reading, not compiled, not run on hardware
//
#include <hip/hip_runtime.h>

// ---------------------------------------------------------------------------
// SpinDistanceEdge on MI455X (gfx1250): per-edge residual MLPs via WMMA f16.
// Each wave32 processes 16 edges (one 16-row WMMA tile). Gaussian 50x50
// residual layers are padded to 64x64 (8x v_wmma_f32_16x16x32_f16 per linear),
// RBF 16x16 layers padded to K=32 (1 WMMA per linear), final 66->16 linear is
// 3 accumulated WMMAs. Weights staged in LDS as f16 [n][k]; activations bounce
// through per-wave LDS tiles between chained matmuls (D-layout -> A-layout).
// ---------------------------------------------------------------------------

typedef __attribute__((ext_vector_type(16))) _Float16 v16h;
typedef __attribute__((ext_vector_type(8)))  _Float16 v8h;
typedef __attribute__((ext_vector_type(8)))  float    v8f;

#define WAVES 4
#define PI_F 3.14159265358979f

#define WMMA_F16(A,B,C) \
  __builtin_amdgcn_wmma_f32_16x16x32_f16(false,(A),false,(B),(short)0,(C),false,false)

__device__ __forceinline__ void wave_sync() {
  // In-order LDS pipeline gives HW ordering within a wave; this pins the
  // compiler's memory-op ordering for cross-lane LDS communication.
  __builtin_amdgcn_wave_barrier();
}

__device__ __forceinline__ float fast_silu(float x) {
  return x * __builtin_amdgcn_rcpf(1.0f + __expf(-x));
}

// A-fragment (16x32 f16 tile): lane m holds K = kc*32 + 16*(lane>=16) .. +15,
// where kb = kc*32 + (lane>=16 ? 8 : 0). Two contiguous 16B LDS loads.
__device__ __forceinline__ v16h load_A_frag(const _Float16* row, int kb) {
  const int hh = (threadIdx.x >> 4) & 1;
  const v8h lo = *(const v8h*)(row + kb + 8 * hh);
  const v8h hi = *(const v8h*)(row + kb + 8 * hh + 8);
  v16h a;
#pragma unroll
  for (int t = 0; t < 8; ++t) { a[t] = lo[t]; a[t + 8] = hi[t]; }
  return a;
}

// B-fragment (32x16 f16 tile): lane holds 16 consecutive K values of one
// output column n; weights staged in LDS as W[n][k] so this is contiguous.
__device__ __forceinline__ v16h load_B_frag(const _Float16* p) {
  const v8h lo = *(const v8h*)p;
  const v8h hi = *(const v8h*)(p + 8);
  v16h b;
#pragma unroll
  for (int t = 0; t < 8; ++t) { b[t] = lo[t]; b[t + 8] = hi[t]; }
  return b;
}

__global__ __launch_bounds__(WAVES * 32)
void spin_edge_kernel(const float* __restrict__ dist,
                      const float* __restrict__ mag,
                      const int*   __restrict__ ei,
                      const int*   __restrict__ ej,
                      const float* __restrict__ linW,
                      const float* __restrict__ linb,
                      const float* __restrict__ rW1, const float* __restrict__ rb1,
                      const float* __restrict__ rW2, const float* __restrict__ rb2,
                      const float* __restrict__ gW1, const float* __restrict__ gb1,
                      const float* __restrict__ gW2, const float* __restrict__ gb2,
                      float* __restrict__ out, int E)
{
  // ---- block-shared staged weights (f16, zero padded) -----------------
  __shared__ _Float16 sGW[2][2][64][64];   // gaussian residual weights [layer][W1/W2][n][k]
  __shared__ _Float16 sRW[2][2][16][32];   // rbf residual weights (K padded 16->32)
  __shared__ _Float16 sFR[16][32];         // final linear, rbf slice   (k 0..15)
  __shared__ _Float16 sFG[16][64];         // final linear, gau slice   (k 0..49)
  __shared__ float    sGB[2][2][64];
  __shared__ float    sRB[2][2][16];
  __shared__ float    sLB[16];
  // ---- per-wave activation tiles --------------------------------------
  __shared__ _Float16 sX[WAVES][16][64];
  __shared__ _Float16 sH[WAVES][16][64];

  const int tid = threadIdx.x;
  const int NT  = WAVES * 32;

  // stage gaussian weights: pad 50x50 -> 64x64
  for (int idx = tid; idx < 2 * 2 * 64 * 64; idx += NT) {
    int k = idx & 63, n = (idx >> 6) & 63, w = (idx >> 12) & 1, l = (idx >> 13) & 1;
    const float* src = w ? gW2 : gW1;
    float v = (n < 50 && k < 50) ? src[(l * 50 + n) * 50 + k] : 0.0f;
    sGW[l][w][n][k] = (_Float16)v;
  }
  // stage rbf weights: pad K 16 -> 32
  for (int idx = tid; idx < 2 * 2 * 16 * 32; idx += NT) {
    int k = idx & 31, n = (idx >> 5) & 15, w = (idx >> 9) & 1, l = (idx >> 10) & 1;
    const float* src = w ? rW2 : rW1;
    float v = (k < 16) ? src[(l * 16 + n) * 16 + k] : 0.0f;
    sRW[l][w][n][k] = (_Float16)v;
  }
  // final linear: W is [16][66]; cols 0..15 -> rbf slice, 16..65 -> gau slice
  for (int idx = tid; idx < 16 * 32; idx += NT) {
    int k = idx & 31, n = idx >> 5;
    sFR[n][k] = (_Float16)((k < 16) ? linW[n * 66 + k] : 0.0f);
  }
  for (int idx = tid; idx < 16 * 64; idx += NT) {
    int k = idx & 63, n = idx >> 6;
    sFG[n][k] = (_Float16)((k < 50) ? linW[n * 66 + 16 + k] : 0.0f);
  }
  for (int idx = tid; idx < 2 * 2 * 64; idx += NT) {
    int n = idx & 63, w = (idx >> 6) & 1, l = (idx >> 7) & 1;
    const float* src = w ? gb2 : gb1;
    sGB[l][w][n] = (n < 50) ? src[l * 50 + n] : 0.0f;
  }
  for (int idx = tid; idx < 2 * 2 * 16; idx += NT) {
    int n = idx & 15, w = (idx >> 4) & 1, l = (idx >> 5) & 1;
    sRB[l][w][n] = (w ? rb2 : rb1)[l * 16 + n];
  }
  if (tid < 16) sLB[tid] = linb[tid];
  __syncthreads();

  const int wave = tid >> 5;
  const int lane = tid & 31;
  const int m    = lane & 15;        // edge row within tile / output col n
  const int half = lane >> 4;        // upper/lower half-wave of A/B striping
  const long eb  = ((long)blockIdx.x * WAVES + wave) * 16;
  long e = eb + m; if (e >= (long)E) e = (long)E - 1;   // clamp (stores guarded)

  _Float16 (*X)[64] = sX[wave];
  _Float16 (*H)[64] = sH[wave];

  // ================= radial Bessel basis -> A fragment + LDS ============
  const float d  = dist[e] * 0.2f;                 // dist / cutoff
  const float x5 = d * d * d * d * d;
  // PyG envelope, p = 6: 1/x - 28 x^5 + 48 x^6 - 21 x^7
  float env = __builtin_amdgcn_rcpf(d) + x5 * (-28.0f + d * (48.0f - 21.0f * d));
  env = (d < 1.0f) ? env : 0.0f;
  {
    const int ks = half * 8;
    v8h lo, hi;
#pragma unroll
    for (int t = 0; t < 8; ++t) {
      lo[t] = (_Float16)(env * __sinf((float)(ks + t + 1) * PI_F * d));
      hi[t] = (_Float16)0.0f;                      // K = 16..31 zero pad
    }
    *(v8h*)(&X[m][ks])      = lo;
    *(v8h*)(&X[m][ks + 16]) = hi;
  }
  wave_sync();

  // ================= 2 residual layers on rbf (16x16, K padded to 32) ===
#pragma unroll
  for (int l = 0; l < 2; ++l) {
    v16h a  = load_A_frag(&X[m][0], half * 8);
    v16h b1f = load_B_frag(&sRW[l][0][m][half * 16]);
    v8f acc = {0.f, 0.f, 0.f, 0.f, 0.f, 0.f, 0.f, 0.f};
    acc = WMMA_F16(a, b1f, acc);
    {
      const float bn = sRB[l][0][m];
#pragma unroll
      for (int r = 0; r < 8; ++r) {
        const int mr = r + 8 * half;               // D row: M = r (+8 upper half)
        H[mr][m]      = (_Float16)fast_silu(acc[r] + bn);
        H[mr][m + 16] = (_Float16)0.0f;            // keep K pad region zero
      }
    }
    wave_sync();
    v16h h   = load_A_frag(&H[m][0], half * 8);
    v16h b2f = load_B_frag(&sRW[l][1][m][half * 16]);
    v8f acc2 = {0.f, 0.f, 0.f, 0.f, 0.f, 0.f, 0.f, 0.f};
    acc2 = WMMA_F16(h, b2f, acc2);
    {
      const float bn = sRB[l][1][m];
#pragma unroll
      for (int r = 0; r < 8; ++r) {
        const int mr = r + 8 * half;
        const float xv = (float)X[mr][m];
        X[mr][m] = (_Float16)(xv + fast_silu(acc2[r] + bn));
      }
    }
    wave_sync();
  }

  // hold final rbf activations in registers through the gaussian phase
  const v16h aR = load_A_frag(&X[m][0], half * 8);
  wave_sync();

  // ================= spin dot + gaussian smearing (pad 50 -> 64) ========
  const int ii = ei[e], jj = ej[e];
  const float* mi = mag + (size_t)ii * 16;
  const float* mj = mag + (size_t)jj * 16;
  float s = 0.0f;
#pragma unroll
  for (int t = 0; t < 16; ++t) s += mi[t] * mj[t];

  const float step  = 5.0f / 49.0f;                // offsets = linspace(0,5,50)
  const float coeff = -0.5f / (step * step);
#pragma unroll
  for (int kc = 0; kc < 2; ++kc) {
    const int kb = kc * 32 + half * 8;
    v8h lo, hi;
#pragma unroll
    for (int t = 0; t < 8; ++t) {
      const int k0 = kb + t, k1 = kb + 16 + t;
      const float d0 = s - (float)k0 * step;
      const float d1 = s - (float)k1 * step;
      lo[t] = (_Float16)((k0 < 50) ? __expf(coeff * d0 * d0) : 0.0f);
      hi[t] = (_Float16)((k1 < 50) ? __expf(coeff * d1 * d1) : 0.0f);
    }
    *(v8h*)(&X[m][kb])      = lo;
    *(v8h*)(&X[m][kb + 16]) = hi;
  }
  wave_sync();

  // ================= 2 residual layers on gaussians (64x64 padded) ======
#pragma unroll
  for (int l = 0; l < 2; ++l) {
    v16h a0 = load_A_frag(&X[m][0], half * 8);
    v16h a1 = load_A_frag(&X[m][0], 32 + half * 8);
#pragma unroll
    for (int nb = 0; nb < 4; ++nb) {
      const int n = nb * 16 + m;
      v16h b0 = load_B_frag(&sGW[l][0][n][half * 16]);
      v16h b1 = load_B_frag(&sGW[l][0][n][32 + half * 16]);
      v8f acc = {0.f, 0.f, 0.f, 0.f, 0.f, 0.f, 0.f, 0.f};
      acc = WMMA_F16(a0, b0, acc);
      acc = WMMA_F16(a1, b1, acc);
      const float bn = sGB[l][0][n];
#pragma unroll
      for (int r = 0; r < 8; ++r)
        H[r + 8 * half][n] = (_Float16)fast_silu(acc[r] + bn);
    }
    wave_sync();
    v16h h0 = load_A_frag(&H[m][0], half * 8);
    v16h h1 = load_A_frag(&H[m][0], 32 + half * 8);
#pragma unroll
    for (int nb = 0; nb < 4; ++nb) {
      const int n = nb * 16 + m;
      v16h b0 = load_B_frag(&sGW[l][1][n][half * 16]);
      v16h b1 = load_B_frag(&sGW[l][1][n][32 + half * 16]);
      v8f acc = {0.f, 0.f, 0.f, 0.f, 0.f, 0.f, 0.f, 0.f};
      acc = WMMA_F16(h0, b0, acc);
      acc = WMMA_F16(h1, b1, acc);
      const float bn = sGB[l][1][n];
#pragma unroll
      for (int r = 0; r < 8; ++r) {
        const int mr = r + 8 * half;
        const float xv = (float)X[mr][n];
        X[mr][n] = (_Float16)(xv + fast_silu(acc[r] + bn));
      }
    }
    wave_sync();
  }

  // ================= final linear: silu([rbf, gau] @ W^T + b) ===========
  v16h g0  = load_A_frag(&X[m][0], half * 8);
  v16h g1  = load_A_frag(&X[m][0], 32 + half * 8);
  v16h br  = load_B_frag(&sFR[m][half * 16]);
  v16h bg0 = load_B_frag(&sFG[m][half * 16]);
  v16h bg1 = load_B_frag(&sFG[m][32 + half * 16]);
  v8f acc = {0.f, 0.f, 0.f, 0.f, 0.f, 0.f, 0.f, 0.f};
  acc = WMMA_F16(aR, br,  acc);
  acc = WMMA_F16(g0, bg0, acc);
  acc = WMMA_F16(g1, bg1, acc);
  const float bn = sLB[m];
#pragma unroll
  for (int r = 0; r < 8; ++r) {
    const int  mr = r + 8 * half;
    const long ee = eb + mr;
    if (ee < (long)E) out[ee * 16 + m] = fast_silu(acc[r] + bn);
  }
}

extern "C" void kernel_launch(void* const* d_in, const int* in_sizes, int n_in,
                              void* d_out, int out_size, void* d_ws, size_t ws_size,
                              hipStream_t stream) {
  const float* dist = (const float*)d_in[0];
  const float* mag  = (const float*)d_in[1];
  const int*   ei   = (const int*)d_in[2];
  const int*   ej   = (const int*)d_in[3];
  // d_in[4] = freq, d_in[5] = offsets: analytic, recomputed on device
  const float* linW = (const float*)d_in[6];
  const float* linb = (const float*)d_in[7];
  const float* rW1  = (const float*)d_in[8];
  const float* rb1  = (const float*)d_in[9];
  const float* rW2  = (const float*)d_in[10];
  const float* rb2  = (const float*)d_in[11];
  const float* gW1  = (const float*)d_in[12];
  const float* gb1  = (const float*)d_in[13];
  const float* gW2  = (const float*)d_in[14];
  const float* gb2  = (const float*)d_in[15];
  float* out = (float*)d_out;

  const int E = in_sizes[0];
  const int edges_per_block = WAVES * 16;          // 64
  const int blocks = (E + edges_per_block - 1) / edges_per_block;

  spin_edge_kernel<<<blocks, WAVES * 32, 0, stream>>>(
      dist, mag, ei, ej, linW, linb,
      rW1, rb1, rW2, rb2, gW1, gb1, gW2, gb2, out, E);
}
